// RandomAttention_23201413333302
// MI455X (gfx1250) — compile-verified, matrix operand order checked by reading, not run on hardware
//
#include <hip/hip_runtime.h>
#include <hip/hip_bf16.h>

typedef __attribute__((ext_vector_type(16))) _Float16 v16h;
typedef __attribute__((ext_vector_type(8)))  float    v8f;

#define B_      2
#define H_      16
#define L_      2048
#define D_      64
#define ROWS    16            // query rows per block (one WMMA M tile)
#define SROW    (L_ + 4)      // padded score row stride (floats): 16B aligned, bank-spread
#define VROW    36            // padded V^T row stride (halves): 4B aligned, bank-spread
#define THREADS 128           // 4 wave32s
#define NFRAG   (L_ / 32)     // 64 K-fragments per GEMM
#define PACK_HALVES ((size_t)B_ * H_ * 4 /*ntiles*/ * NFRAG * 512)   // == B*H*L*D

// K-within-fragment mapping of the 16-bit WMMA A/B VGPR layout:
// half j (0..15) of a lane -> local K offset (before the half-wave +8/+0 split).
__device__ __forceinline__ int frag_k(int j) {
    int p  = j >> 1;
    int kl = (p < 4) ? (2 * p) : (16 + 2 * (p - 4));
    return kl + (j & 1);
}

// ---------------------------------------------------------------------------
// Kernel 1: pack V (f32, [B,H,L,D]) into f16 B-fragment-major layout in ws.
// Fragment (bh, ntile, kf) = 32x16 tile of V: 32 lanes x 16 halves contiguous,
// so the main kernel loads one v16h per lane with a single coalesced access.
// ---------------------------------------------------------------------------
__global__ __launch_bounds__(256)
void pack_v_kernel(const float* __restrict__ V, _Float16* __restrict__ vp)
{
    size_t g = (size_t)blockIdx.x * 256 + threadIdx.x;   // one half per thread
    int j    = (int)(g        & 15);
    int lane = (int)((g >> 4) & 31);
    int kf   = (int)((g >> 9) & (NFRAG - 1));
    int nt   = (int)((g >> 15) & 3);
    int bh   = (int)( g >> 17);

    int ko  = (lane < 16) ? 0 : 8;
    int k   = kf * 32 + ko + frag_k(j);
    int col = nt * 16 + (lane & 15);

    vp[g] = (_Float16)V[((size_t)bh * L_ + k) * D_ + col];
}

// ---------------------------------------------------------------------------
// Kernel 2: fused softmax + (weights @ V) with WMMA.
// PACKED=true : B fragments loaded directly from vp (L2-resident, no barriers)
// PACKED=false: fallback, stage V chunks through LDS per K step.
// ---------------------------------------------------------------------------
template<bool PACKED>
__global__ __launch_bounds__(THREADS)
void rattn_fused(const float* __restrict__ V,
                 const float* __restrict__ attn,
                 const _Float16* __restrict__ vp,
                 float* __restrict__ out,       // [B,H,L,D] f32
                 float* __restrict__ weights)   // [B,H,L,L] f32
{
    extern __shared__ char smem_raw[];
    float*    s_sc   = (float*)smem_raw;                                  // ROWS*SROW f32
    _Float16* s_vt   = (_Float16*)(smem_raw + ROWS * SROW * 4);           // D_*VROW f16
    float*    s_rinv = (float*)(smem_raw + ROWS * SROW * 4 + D_ * VROW * 2); // ROWS f32

    const int tid  = threadIdx.x;
    const int lane = tid & 31;
    const int wave = tid >> 5;

    const int bh = blockIdx.x >> 7;          // 128 q-tiles per (b,h)
    const int qt = blockIdx.x & 127;
    const int q0 = qt * ROWS;

    const float* attn_base = attn + ((size_t)bh * L_ + q0) * L_;
    const float* v_base    = V    +  (size_t)bh * L_ * D_;

    // ---- Phase 1: stage 16 x 2048 f32 scores into LDS (coalesced float4) ----
    #pragma unroll 4
    for (int i = 0; i < (ROWS * L_ / 4) / THREADS; ++i) {   // 64 iters
        int e  = i * THREADS + tid;    // float4 index
        int r  = e >> 9;               // 512 float4 per row
        int c4 = e & 511;
        float4 v4 = ((const float4*)(attn_base + (size_t)r * L_))[c4];
        *(float4*)(s_sc + r * SROW + c4 * 4) = v4;
    }
    __syncthreads();

    // ---- Phase 2: rowwise softmax; store UNNORMALIZED exp back in place ----
    for (int j = 0; j < 4; ++j) {
        int r = wave * 4 + j;
        float* row = s_sc + r * SROW;
        float m = -3.402823466e+38f;
        for (int i = 0; i < L_ / 32; ++i) m = fmaxf(m, row[lane + 32 * i]);
        for (int off = 16; off > 0; off >>= 1) m = fmaxf(m, __shfl_xor(m, off, 32));
        float s = 0.f;
        for (int i = 0; i < L_ / 32; ++i) {
            float e = __expf(row[lane + 32 * i] - m);
            row[lane + 32 * i] = e;     // same lane owns read & write: no race
            s += e;
        }
        for (int off = 16; off > 0; off >>= 1) s += __shfl_xor(s, off, 32);
        if (lane == 0) s_rinv[r] = 1.0f / s;
    }
    __syncthreads();

    // ---- Phase 3: stream normalized f32 weights to global (coalesced) ----
    #pragma unroll 4
    for (int i = 0; i < (ROWS * L_ / 4) / THREADS; ++i) {
        int e  = i * THREADS + tid;
        int r  = e >> 9;
        int c4 = e & 511;
        float4 v4 = *(const float4*)(s_sc + r * SROW + c4 * 4);
        float inv = s_rinv[r];
        v4.x *= inv; v4.y *= inv; v4.z *= inv; v4.w *= inv;
        ((float4*)(weights + ((size_t)bh * L_ + q0 + r) * L_))[c4] = v4;
    }

    // ---- GEMM: C(16x16 per wave) = exp(16x2048) @ V(2048x64) ----
    v8f c = {};
    const int nb   = wave * 16;
    const int mrow = lane & 15;               // A row / B column within tile
    const int ko   = (lane < 16) ? 0 : 8;     // half-wave K offset per WMMA layout

    const v16h* bbase =
        (const v16h*)(vp + ((size_t)(bh * 4 + wave) * NFRAG) * 512);

    for (int kf = 0; kf < NFRAG; ++kf) {
        v16h a, b;

        if constexpr (PACKED) {
            b = bbase[kf * 32 + lane];                    // 32B coalesced per lane
            int pk = (kf + 2 < NFRAG) ? (kf + 2) : (NFRAG - 1);
            __builtin_prefetch(&bbase[pk * 32 + lane], 0, 0);  // global_prefetch
        } else {
            // stage V chunk (32 x 64 f32) transposed -> s_vt[d][k] as f16
            #pragma unroll
            for (int i = 0; i < 4; ++i) {
                int e  = i * THREADS + tid;
                int k  = e >> 4;
                int c4 = e & 15;
                float4 vv = ((const float4*)(v_base + (size_t)(kf * 32 + k) * D_))[c4];
                s_vt[(c4 * 4 + 0) * VROW + k] = (_Float16)vv.x;
                s_vt[(c4 * 4 + 1) * VROW + k] = (_Float16)vv.y;
                s_vt[(c4 * 4 + 2) * VROW + k] = (_Float16)vv.z;
                s_vt[(c4 * 4 + 3) * VROW + k] = (_Float16)vv.w;
            }
            __syncthreads();
            const _Float16* bcol = s_vt + (nb + mrow) * VROW + ko;
            #pragma unroll
            for (int j = 0; j < 16; ++j) b[j] = bcol[frag_k(j)];
        }

        const float* arow = s_sc + mrow * SROW + kf * 32 + ko;
        #pragma unroll
        for (int v = 0; v < 8; ++v) {
            int kl = (v < 4) ? (2 * v) : (16 + 2 * (v - 4));
            a[2 * v]     = (_Float16)arow[kl];
            a[2 * v + 1] = (_Float16)arow[kl + 1];
        }

        c = __builtin_amdgcn_wmma_f32_16x16x32_f16(false, a, false, b,
                                                   (short)0, c, false, false);
        if constexpr (!PACKED) __syncthreads();   // protect s_vt before restage
    }

    // ---- Epilogue: fold 1/rowsum into C, store output ----
    const int hi = lane >> 4;   // C layout: VGPR i holds M = i + 8*hi, N = lane&15
    #pragma unroll
    for (int i = 0; i < 8; ++i) {
        int m = i + 8 * hi;
        float o = c[i] * s_rinv[m];
        out[((size_t)bh * L_ + q0 + m) * D_ + nb + mrow] = o;
    }
}

extern "C" void kernel_launch(void* const* d_in, const int* in_sizes, int n_in,
                              void* d_out, int out_size, void* d_ws, size_t ws_size,
                              hipStream_t stream) {
    // inputs: Q(unused), K(unused), V, random_attn, len_q(unused; fixed L=2048)
    const float* V    = (const float*)d_in[2];
    const float* attn = (const float*)d_in[3];
    float* out        = (float*)d_out;
    float* weights    = out + (size_t)B_ * H_ * L_ * D_;

    const size_t smem = (size_t)ROWS * SROW * 4   // scores / exp tile
                      + (size_t)D_ * VROW * 2     // V^T f16 chunk (fallback path)
                      + (size_t)ROWS * 4;         // 1/rowsum

    dim3 grid(B_ * H_ * (L_ / ROWS));   // 4096 blocks
    dim3 block(THREADS);                // 4 wave32s

    const size_t need = PACK_HALVES * sizeof(_Float16);   // 8 MB
    if (ws_size >= need && d_ws != nullptr) {
        _Float16* vp = (_Float16*)d_ws;
        (void)hipFuncSetAttribute((const void*)rattn_fused<true>,
                                  hipFuncAttributeMaxDynamicSharedMemorySize, (int)smem);
        pack_v_kernel<<<dim3((unsigned)(PACK_HALVES / 256)), dim3(256), 0, stream>>>(V, vp);
        rattn_fused<true><<<grid, block, smem, stream>>>(V, attn, vp, out, weights);
    } else {
        (void)hipFuncSetAttribute((const void*)rattn_fused<false>,
                                  hipFuncAttributeMaxDynamicSharedMemorySize, (int)smem);
        rattn_fused<false><<<grid, block, smem, stream>>>(V, attn, nullptr, out, weights);
    }
}
